// StyleLinear_44186623541957
// MI455X (gfx1250) — matrix operand order repaired and checked by reading, not compiled, not run on hardware
//
#include <hip/hip_runtime.h>
#include <hip/hip_bf16.h>

typedef unsigned short u16;
typedef __attribute__((ext_vector_type(16))) __bf16 v16bf;
typedef __attribute__((ext_vector_type(16))) u16    v16u;
typedef __attribute__((ext_vector_type(8)))  float  v8f;
typedef __attribute__((ext_vector_type(4)))  float  v4f;

__device__ __forceinline__ u16 f32_to_bf16_rne(float f) {
  unsigned int u = __float_as_uint(f);
  u += 0x7fffu + ((u >> 16) & 1u);
  return (u16)(u >> 16);
}
__device__ __forceinline__ float bf16_to_f32(u16 h) {
  return __uint_as_float(((unsigned int)h) << 16);
}

// Async DMA of 64 bytes (one wmod row slice) straight into LDS, no VGPR
// round-trip. Tracked by ASYNCcnt. LDS operand is the wave-relative byte
// offset (low 32 bits of the flat shared address).
__device__ __forceinline__ void async_stage_row(const u16* gsrc, u16* ldst) {
  #pragma unroll
  for (int q = 0; q < 4; ++q) {
    unsigned           l = (unsigned)(uintptr_t)(ldst + q * 8);
    unsigned long long g = (unsigned long long)(uintptr_t)(gsrc + q * 8);
    asm volatile("global_load_async_to_lds_b128 %0, %1, off"
                 :: "v"(l), "v"(g) : "memory");
  }
}

// ---------------------------------------------------------------------------
// Kernel 1: styles = w @ affine_w^T + affine_b ; modulate+demodulate weight;
// write bf16 hi/lo split of wmod[n][o][i] to workspace.
// grid: 16 blocks (samples), 256 threads (output channels)
// ---------------------------------------------------------------------------
__global__ __launch_bounds__(256) void style_modw_kernel(
    const float* __restrict__ w, const float* __restrict__ weight,
    const float* __restrict__ aff_w, const float* __restrict__ aff_b,
    u16* __restrict__ bhi, u16* __restrict__ blo)
{
  const int n = blockIdx.x;
  const int o = threadIdx.x;
  __shared__ float st[8];
  if (o < 8) {
    float s = aff_b[o];
    const float* wr = w + n * 512;
    const float* ar = aff_w + o * 512;
    for (int i = 0; i < 512; ++i) s += wr[i] * ar[i];
    st[o] = s;
  }
  __syncthreads();
  float sl[8];
  #pragma unroll
  for (int g = 0; g < 8; ++g) sl[g] = st[g];

  const float* wr = weight + o * 256;
  float ss = 0.f;
  for (int i = 0; i < 256; ++i) { float v = wr[i] * sl[i >> 5]; ss += v * v; }
  const float d = rsqrtf(ss + 1e-8f);

  u16* ph = bhi + ((size_t)(n * 256 + o)) * 256;
  u16* pl = blo + ((size_t)(n * 256 + o)) * 256;
  for (int i = 0; i < 256; ++i) {
    float v = wr[i] * sl[i >> 5] * d;
    u16 h = f32_to_bf16_rne(v);
    ph[i] = h;
    pl[i] = f32_to_bf16_rne(v - bf16_to_f32(h));
  }
}

// ---------------------------------------------------------------------------
// Kernel 2: out[n,hw,o] = sum_i x[n,hw,i] * wmod[n,o,i] + bias[o]
// bf16x3 split-precision GEMM on v_wmma_f32_16x16x32_bf16.
// grid: (4096/128, 16) blocks, 256 threads = 8 waves.
// Each wave: M-slab of 16 rows, all 16 N-tiles (16x16), K-loop step 32.
// B slab (hi+lo) double-buffered in LDS via async-to-LDS DMA.
// ---------------------------------------------------------------------------
__global__ __launch_bounds__(256) void style_gemm_kernel(
    const float* __restrict__ x, const u16* __restrict__ bhi,
    const u16* __restrict__ blo, const float* __restrict__ bias,
    float* __restrict__ out)
{
  __shared__ __align__(32) u16 sBhi[2][256 * 32];
  __shared__ __align__(32) u16 sBlo[2][256 * 32];

  const int n    = blockIdx.y;
  const int m0   = blockIdx.x * 128;
  const int tid  = threadIdx.x;
  const int lane = tid & 31;
  const int wv   = tid >> 5;
  const int half = lane >> 4;   // which 16-lane half of the wave
  const int l16  = lane & 15;

  // A fragment rows: lane%16 selects row within wave's 16-row slab.
  const float* xr = x + ((size_t)((n << 12) + m0 + (wv << 4) + l16)) * 256;
  const int kb = half << 4;     // contiguous K layout: halves own K 0-15 vs 16-31

  v8f zero = {};
  v8f acc[16];
  #pragma unroll
  for (int t = 0; t < 16; ++t) acc[t] = zero;

  const u16* gh = bhi + ((size_t)(n * 256 + tid)) * 256;
  const u16* gl = blo + ((size_t)(n * 256 + tid)) * 256;

  // Prologue: DMA first K-slab into buffer 0.
  async_stage_row(gh, &sBhi[0][tid * 32]);
  async_stage_row(gl, &sBlo[0][tid * 32]);

  for (int k0 = 0; k0 < 256; k0 += 32) {
    const int cur = (k0 >> 5) & 1;
    if (k0 + 32 < 256) {
      // Issue next slab into the other buffer, then drain the current one
      // (async loads complete in order: <=8 outstanding == current done).
      async_stage_row(gh + k0 + 32, &sBhi[cur ^ 1][tid * 32]);
      async_stage_row(gl + k0 + 32, &sBlo[cur ^ 1][tid * 32]);
      asm volatile("s_wait_asynccnt 0x8" ::: "memory");
    } else {
      asm volatile("s_wait_asynccnt 0x0" ::: "memory");
    }
    __syncthreads();

    // Load 16 fp32 of x for this lane (streamed once -> non-temporal) and
    // split into bf16 hi/lo fragments.
    const float* xk = xr + k0 + kb;
    v4f f0 = __builtin_nontemporal_load((const v4f*)(xk + 0));
    v4f f1 = __builtin_nontemporal_load((const v4f*)(xk + 4));
    v4f f2 = __builtin_nontemporal_load((const v4f*)(xk + 8));
    v4f f3 = __builtin_nontemporal_load((const v4f*)(xk + 12));
    float fa[16] = { f0.x, f0.y, f0.z, f0.w, f1.x, f1.y, f1.z, f1.w,
                     f2.x, f2.y, f2.z, f2.w, f3.x, f3.y, f3.z, f3.w };
    v16u ahu, alu;
    #pragma unroll
    for (int i = 0; i < 16; ++i) {
      u16 h = f32_to_bf16_rne(fa[i]);
      ahu[i] = h;
      alu[i] = f32_to_bf16_rne(fa[i] - bf16_to_f32(h));
    }
    v16bf ah = __builtin_bit_cast(v16bf, ahu);
    v16bf al = __builtin_bit_cast(v16bf, alu);

    // 16 N-tiles, 3 WMMAs each (hi*hi + hi*lo + lo*hi), fp32 accumulate.
    #pragma unroll
    for (int t = 0; t < 16; ++t) {
      const int off = ((t * 16 + l16) * 32) + (half << 4); // B: N=lane%16, K-half=lane/16
      v16bf bh = __builtin_bit_cast(v16bf, *(const v16u*)(&sBhi[cur][off]));
      v16bf bl = __builtin_bit_cast(v16bf, *(const v16u*)(&sBlo[cur][off]));
      acc[t] = __builtin_amdgcn_wmma_f32_16x16x32_bf16(
          false, ah, false, bh, (short)0, acc[t], false, false);
      acc[t] = __builtin_amdgcn_wmma_f32_16x16x32_bf16(
          false, ah, false, bl, (short)0, acc[t], false, false);
      acc[t] = __builtin_amdgcn_wmma_f32_16x16x32_bf16(
          false, al, false, bh, (short)0, acc[t], false, false);
    }
    __syncthreads();
  }

  // C/D layout: lane half selects M offset 0/8, VGPR j is row j within that.
  // out is streamed once -> non-temporal stores keep wmod slabs hot in L2.
  const int rbase = (n << 12) + m0 + (wv << 4) + (half << 3);
  #pragma unroll
  for (int t = 0; t < 16; ++t) {
    const int o = t * 16 + l16;
    const float bs = bias[o];
    #pragma unroll
    for (int j = 0; j < 8; ++j) {
      __builtin_nontemporal_store(acc[t][j] + bs,
                                  &out[((size_t)(rbase + j)) * 256 + o]);
    }
  }
}

extern "C" void kernel_launch(void* const* d_in, const int* in_sizes, int n_in,
                              void* d_out, int out_size, void* d_ws, size_t ws_size,
                              hipStream_t stream) {
  (void)in_sizes; (void)n_in; (void)out_size; (void)ws_size;
  const float* x      = (const float*)d_in[0];
  const float* w      = (const float*)d_in[1];
  const float* weight = (const float*)d_in[2];
  const float* bias   = (const float*)d_in[3];
  const float* aw     = (const float*)d_in[4];
  const float* ab     = (const float*)d_in[5];
  float* out = (float*)d_out;

  u16* bhi = (u16*)d_ws;                          // 16*256*256 u16 = 2 MB
  u16* blo = bhi + (size_t)16 * 256 * 256;        // +2 MB

  style_modw_kernel<<<dim3(16), dim3(256), 0, stream>>>(w, weight, aw, ab, bhi, blo);
  style_gemm_kernel<<<dim3(32, 16), dim3(256), 0, stream>>>(x, bhi, blo, bias, out);
}
